// GaussianBlurFault_33371895890241
// MI455X (gfx1250) — compile-verified, matrix operand order checked by reading, not run on hardware
//
#include <hip/hip_runtime.h>
#include <hip/hip_bf16.h>
#include <math.h>

typedef _Float16 f16;
typedef __attribute__((ext_vector_type(16))) _Float16 v16h;
typedef __attribute__((ext_vector_type(8)))  _Float16 v8h;
typedef __attribute__((ext_vector_type(4)))  _Float16 v4h;
typedef __attribute__((ext_vector_type(8)))  float    v8f;
typedef __attribute__((ext_vector_type(4)))  float    v4f;

#define IMG       512
#define KTAP      25
#define GPAD      12
#define SIGMA2    (2.0f * 9.0f * 9.0f)

#define TILE      64          // output tile edge
#define IN_DIM    88          // TILE + 2*GPAD
#define IN_ROWS   96          // padded to 6 groups of 16 rows
#define IN_STRIDE 112         // padded cols so all b128 loads are in-bounds (224B row, 16B-mult)
#define H_COLS    64
#define H_ROWS    112         // transposed-H row dimension (stride), zero-tail 96..111

static __device__ __forceinline__ v16h cat16(v8h lo, v8h hi) {
    return __builtin_shufflevector(lo, hi, 0,1,2,3,4,5,6,7,8,9,10,11,12,13,14,15);
}

__global__ __launch_bounds__(256)
void gaussblur_wmma_kernel(const float* __restrict__ x, float* __restrict__ y) {
    __shared__ __align__(16) f16 in_s[IN_ROWS * IN_STRIDE];  // input tile, f16, zero-padded
    __shared__ __align__(16) f16 h_t [H_COLS  * H_ROWS];     // horizontal result, transposed
    // Banded Toeplitz weight matrices, pre-laid-out in WMMA register order:
    // row = lane (0..31), col = element (0..15). W1 = K-tile 0, W2 = K-tile 1 (k+32).
    __shared__ __align__(16) f16 w1_s[32 * 16];
    __shared__ __align__(16) f16 w2_s[32 * 16];

    const int tid  = threadIdx.x;
    const int lane = tid & 31;
    const int wave = tid >> 5;
    const int half = (lane >> 4) & 1;   // which 16-lane half of the wave
    const int l16  = lane & 15;

    const int C0 = blockIdx.x * TILE;
    const int R0 = blockIdx.y * TILE;
    const size_t planeOff = (size_t)blockIdx.z * (size_t)(IMG * IMG);
    const float* __restrict__ xp = x + planeOff;

    // ---- per-thread Gaussian normalization (registers only) ----
    float gsum = 0.f;
    #pragma unroll
    for (int i = 0; i < KTAP; ++i) {
        float d = (float)(i - GPAD);
        gsum += __expf(-d * d / SIGMA2);
    }
    const float ginv = 1.0f / gsum;

    // ---- build weight tables once (4 entries per thread) ----
    // element->K map (wave32 WMMA 16-bit layout): k = 8*half + e + (e>=8 ? 8 : 0)
    for (int idx = tid; idx < 2 * 32 * 16; idx += 256) {
        int tab = idx >> 9;                 // 0 -> W1, 1 -> W2
        int j   = idx & 511;
        int L   = j >> 4, e = j & 15;
        int hlf = (L >> 4) & 1, c = L & 15;
        int k   = 8 * hlf + e + ((e >= 8) ? 8 : 0) + tab * 32;
        int t   = k - c;
        float val = 0.f;
        if (t >= 0 && t < KTAP) {
            float d = (float)(t - GPAD);
            val = __expf(-d * d / SIGMA2) * ginv;
        }
        (tab ? w2_s : w1_s)[j] = (f16)val;
    }

    const v8h z8 = {};

    // ---- cooperative halo'd tile load, f32 -> f16 ----
    // Interior blocks (uniform branch): vectorized float4 path, no per-pixel clamps.
    const bool interior = (R0 >= GPAD) && (R0 + TILE + GPAD <= IMG) &&
                          (C0 >= GPAD) && (C0 + TILE + GPAD <= IMG);
    if (interior) {
        const float* base = xp + (size_t)(R0 - GPAD) * IMG + (C0 - GPAD); // 16B-aligned rows
        for (int idx = tid; idx < IN_DIM * 22; idx += 256) {   // 88 rows x 22 float4
            int r = idx / 22, q = idx - r * 22;
            v4f v = *(const v4f*)(base + (size_t)r * IMG + 4 * q);
            v4h h;
            #pragma unroll
            for (int i = 0; i < 4; ++i) h[i] = (f16)v[i];
            *(v4h*)&in_s[r * IN_STRIDE + 4 * q] = h;
        }
        // zero-fill padding cols 88..111 (rows 0..87): 3 x b128 per row
        for (int idx = tid; idx < IN_DIM * 3; idx += 256) {
            int r = idx / 3, q = idx - r * 3;
            *(v8h*)&in_s[r * IN_STRIDE + 88 + 8 * q] = z8;
        }
        // zero-fill padding rows 88..95 (full width): 14 x b128 per row
        for (int idx = tid; idx < 8 * 14; idx += 256) {
            int r = idx / 14, q = idx - r * 14;
            *(v8h*)&in_s[(88 + r) * IN_STRIDE + 8 * q] = z8;
        }
    } else {
        // Edge blocks: fully clamped scalar path (zero padding outside the image)
        for (int idx = tid; idx < IN_ROWS * IN_STRIDE; idx += 256) {
            int r = idx / IN_STRIDE, c = idx % IN_STRIDE;
            int gr = R0 - GPAD + r, gc = C0 - GPAD + c;
            float v = 0.f;
            if (r < IN_DIM && c < IN_DIM && gr >= 0 && gr < IMG && gc >= 0 && gc < IMG)
                v = xp[gr * IMG + gc];
            in_s[idx] = (f16)v;
        }
    }
    // zero tail rows 96..111 of transposed H (read by stage-2 K-tile 2 under zero weights)
    for (int idx = tid; idx < H_COLS * 2; idx += 256) {
        int c = idx >> 1, q = idx & 1;
        *(v8h*)&h_t[c * H_ROWS + 96 + 8 * q] = z8;
    }
    __syncthreads();

    // ---- load banded weights once per lane; shared by stage 1 (as B) and stage 2 (as A) ----
    const f16* wr1 = &w1_s[lane * 16];
    const f16* wr2 = &w2_s[lane * 16];
    const v16h W1 = cat16(*(const v8h*)wr1, *(const v8h*)(wr1 + 8));
    const v16h W2 = cat16(*(const v8h*)wr2, *(const v8h*)(wr2 + 8));

    // ================= Stage 1: horizontal blur, H = In * Wband =================
    // 24 subtiles (6 row-groups x 4 col-groups) over 8 waves, 3 each.
    for (int it = 0; it < 3; ++it) {
        int s  = wave + 8 * it;          // 0..23
        int rg = s >> 2;                 // H row group 0..5 (rows rg*16..rg*16+15)
        int g0 = (s & 3) << 4;           // output-col group base 0/16/32/48
        const f16* arow = &in_s[(rg * 16 + l16) * IN_STRIDE];

        // A tiles: A1[m][k] = in_s[row][g0 + k], A2 = cols g0+32+k (contiguous per 8-elem half)
        v16h A1 = cat16(*(const v8h*)(arow + g0 +      8 * half),
                        *(const v8h*)(arow + g0 + 16 + 8 * half));
        v16h A2 = cat16(*(const v8h*)(arow + g0 + 32 + 8 * half),
                        *(const v8h*)(arow + g0 + 48 + 8 * half));

        v8f C = {};
        C = __builtin_amdgcn_wmma_f32_16x16x32_f16(false, A1, false, W1, (short)0, C, false, false);
        C = __builtin_amdgcn_wmma_f32_16x16x32_f16(false, A2, false, W2, (short)0, C, false, false);

        // store transposed: h_t[col][row]; lane holds col g0+l16, rows rg*16+8*half+v (v=0..7)
        v8h hv;
        #pragma unroll
        for (int v = 0; v < 8; ++v) hv[v] = (f16)C[v];
        *(v8h*)&h_t[(g0 + l16) * H_ROWS + rg * 16 + 8 * half] = hv;
    }
    __syncthreads();

    // ================= Stage 2: vertical blur, Out = Wband * H =================
    // 16 subtiles (4 row-groups x 4 col-groups) over 8 waves, 2 each.
    for (int it = 0; it < 2; ++it) {
        int s  = wave + 8 * it;          // 0..15
        int mg = (s >> 2) << 4;          // out row group base 0/16/32/48
        int g0 = (s & 3) << 4;           // out col group base

        // B tiles: H data, B1[k][c] = h_t[g0+c][mg+k], B2 = rows mg+32+k (contiguous)
        const f16* hcol = &h_t[(g0 + l16) * H_ROWS];
        v16h B1 = cat16(*(const v8h*)(hcol + mg +      8 * half),
                        *(const v8h*)(hcol + mg + 16 + 8 * half));
        v16h B2 = cat16(*(const v8h*)(hcol + mg + 32 + 8 * half),
                        *(const v8h*)(hcol + mg + 48 + 8 * half));

        v8f C = {};
        C = __builtin_amdgcn_wmma_f32_16x16x32_f16(false, W1, false, B1, (short)0, C, false, false);
        C = __builtin_amdgcn_wmma_f32_16x16x32_f16(false, W2, false, B2, (short)0, C, false, false);

        // clip + store: rows R0+mg+8*half+v, col C0+g0+l16
        float* yp = y + planeOff + (size_t)(R0 + mg + 8 * half) * IMG + (C0 + g0 + l16);
        #pragma unroll
        for (int v = 0; v < 8; ++v) {
            float o = C[v];
            o = fminf(fmaxf(o, 0.f), 1.f);
            yp[(size_t)v * IMG] = o;
        }
    }
}

extern "C" void kernel_launch(void* const* d_in, const int* in_sizes, int n_in,
                              void* d_out, int out_size, void* d_ws, size_t ws_size,
                              hipStream_t stream) {
    (void)n_in; (void)out_size; (void)d_ws; (void)ws_size;
    const float* x = (const float*)d_in[0];
    float* y = (float*)d_out;
    int planes = in_sizes[0] / (IMG * IMG);   // N * C = 96
    dim3 grid(IMG / TILE, IMG / TILE, planes);
    gaussblur_wmma_kernel<<<grid, dim3(256), 0, stream>>>(x, y);
}
